// DotInteraction_86792699118126
// MI455X (gfx1250) — compile-verified, matrix operand order checked by reading, not run on hardware
//
#include <hip/hip_runtime.h>
#include <hip/hip_bf16.h>

typedef __attribute__((ext_vector_type(16))) __bf16 bf16x16;
typedef __attribute__((ext_vector_type(8)))  __bf16 bf16x8;
typedef __attribute__((ext_vector_type(8)))  float  f32x8;

#define BT      32        // batch rows per workgroup
#define DFEAT   128
#define XLOG    2176      // logical K for MLP1: 120 pairwise + 2048 flat + 8 zero pad
#define XSTR    2184      // storage stride (+16B) -> conflict-free LDS banking
#define H1LOG   1024
#define H1STR   1032
#define H2LOG   512
#define H2STR   520
#define OUTLEN  256
#define KS1     (XLOG / 32)   // 68
#define KS2     (H1LOG / 32)  // 32
#define KS3     (H2LOG / 32)  // 16
#define KSP     2             // projection K = 64

struct Params {
  const float* f[16];
  const float* bp;
  const float* b1;
  const float* b2;
  const float* b3;
  const __bf16* WpP;   // packed frags: 8 nt x 2 ks x 512
  const __bf16* W1P;   // 64 nt x 68 ks x 512
  const __bf16* W2P;   // 32 nt x 32 ks x 512
  const __bf16* W3P;   // 16 nt x 16 ks x 512
  float* out;          // B x 256
};

// ---- A-fragment loader: CDNA5 16-bit 16x32 layout (lane m = row; two
// contiguous 8-element K chunks -> two 16B loads per lane) ----
__device__ __forceinline__ bf16x16 load_frag(const __bf16* base, int stride, int lane) {
  const int m = lane & 15, h = (lane >> 4) & 1;
  const __bf16* p = base + m * stride + h * 8;
  bf16x8 lo = *(const bf16x8*)(p);        // K = 8h .. 8h+7
  bf16x8 hi = *(const bf16x8*)(p + 16);   // K = 16+8h .. 16+8h+7
  return __builtin_shufflevector(lo, hi, 0,1,2,3,4,5,6,7,8,9,10,11,12,13,14,15);
}

// ---- B-fragment loader from pre-packed weights: 32B contiguous per lane ----
__device__ __forceinline__ bf16x16 load_frag_pk(const __bf16* frag, int lane) {
  return *(const bf16x16*)(frag + lane * 16);
}

__device__ __forceinline__ bf16x16 load_frag_f32(const float* base, int stride, int lane) {
  const int m = lane & 15, h = (lane >> 4) & 1;
  const float* p = base + m * stride + h * 8;
  bf16x16 v;
#pragma unroll
  for (int i = 0; i < 8; ++i) { v[i] = (__bf16)p[i]; v[i + 8] = (__bf16)p[16 + i]; }
  return v;
}

__device__ __forceinline__ f32x8 wmma_bf16(bf16x16 a, bf16x16 b, f32x8 c) {
  return __builtin_amdgcn_wmma_f32_16x16x32_bf16(false, a, false, b, (short)0, c, false, false);
}

// ---- weight prep: f32 (K x N) -> bf16 WMMA-B fragment order, zero-padded K ----
// dst layout: [nt][ks][lane][16 elems]; element i of lane maps to
// K = ks*32 + (i<8 ? 8h+i : 16+8h+(i-8)),  N = nt*16 + (lane&15)
__global__ void weight_pack_bf16(const float* __restrict__ src, __bf16* __restrict__ dst,
                                 int K, int N, int ksteps) {
  int idx = blockIdx.x * blockDim.x + threadIdx.x;
  int total = (N / 16) * ksteps * 512;
  if (idx >= total) return;
  int i    = idx & 15;
  int lane = (idx >> 4) & 31;
  int fs   = idx >> 9;
  int ks   = fs % ksteps;
  int nt   = fs / ksteps;
  int n = nt * 16 + (lane & 15);
  int h = (lane >> 4) & 1;
  int k = ks * 32 + ((i < 8) ? (8 * h + i) : (16 + 8 * h + (i - 8)));
  dst[idx] = (k < K) ? (__bf16)src[(size_t)k * N + n] : (__bf16)0.0f;
}

// ---- fused: projection + stack + self-dot interaction + 3-layer MLP ----
extern "C" __global__ void __launch_bounds__(256)
fused_dlrm(Params prm) {
  extern __shared__ __align__(16) char smem[];
  __bf16* sX  = (__bf16*)smem;                                   // BT x XSTR
  __bf16* sH1 = (__bf16*)(smem + (size_t)BT * XSTR * 2);         // BT x H1STR
  __bf16* sH2 = (__bf16*)(smem + (size_t)BT * XSTR * 2 + (size_t)BT * H1STR * 2);

  const int tid  = threadIdx.x;
  const int lane = tid & 31;
  const int wave = tid >> 5;
  const int b0   = blockIdx.x * BT;

  // ---- Stage 1a: features f1..f15 -> bf16 flat region of x ----
#pragma unroll
  for (int fi = 1; fi <= 15; ++fi) {
    const float* src = prm.f[fi];
    for (int idx = tid; idx < BT * DFEAT / 4; idx += 256) {
      int r = idx >> 5;            // 32 float4 per row
      int d4 = idx & 31;
      float4 v = ((const float4*)(src + (size_t)(b0 + r) * DFEAT))[d4];
      __bf16* dst = sX + (size_t)r * XSTR + 120 + fi * DFEAT + d4 * 4;
      dst[0] = (__bf16)v.x; dst[1] = (__bf16)v.y;
      dst[2] = (__bf16)v.z; dst[3] = (__bf16)v.w;
    }
  }
  { // zero pad elements 2168..2175 of every row
    int r = tid >> 3, j = tid & 7;
    sX[(size_t)r * XSTR + 2168 + j] = (__bf16)0.0f;
  }

  // ---- Stage 1b: projection p0 = f0 @ Wp + bp -> flat slot 0 ----
  // 2 Mtiles x 8 Ntiles; each wave: 1 Mtile x 2 Ntiles (A reused across N)
  {
    const int mt = wave >> 2, ntg = wave & 3;           // nt = ntg*2 + j
    f32x8 acc[2] = {};
#pragma unroll
    for (int ks = 0; ks < KSP; ++ks) {
      bf16x16 a = load_frag_f32(prm.f[0] + (size_t)(b0 + mt * 16) * 64 + ks * 32, 64, lane);
#pragma unroll
      for (int j = 0; j < 2; ++j) {
        const int nt = ntg * 2 + j;
        bf16x16 b = load_frag_pk(prm.WpP + (size_t)(nt * KSP + ks) * 512, lane);
        acc[j] = wmma_bf16(a, b, acc[j]);
      }
    }
    const int mbase = mt * 16 + ((lane >> 4) & 1) * 8;
#pragma unroll
    for (int j = 0; j < 2; ++j) {
      const int n = (ntg * 2 + j) * 16 + (lane & 15);
      const float bias = prm.bp[n];
#pragma unroll
      for (int r = 0; r < 8; ++r)
        sX[(size_t)(mbase + r) * XSTR + 120 + n] = (__bf16)(acc[j][r] + bias);
    }
  }
  __syncthreads();

  // ---- Stage 2: dots = S_b @ S_b^T per sample; scatter upper triangle ----
#pragma unroll
  for (int s = wave * 4; s < wave * 4 + 4; ++s) {
    const __bf16* Sb = sX + (size_t)s * XSTR + 120;  // 16 x 128 row-major
    f32x8 acc = {};
#pragma unroll
    for (int ks = 0; ks < 4; ++ks) {
      bf16x16 a = load_frag(Sb + ks * 32, DFEAT, lane);
      acc = wmma_bf16(a, a, acc);   // B from same tile: B(k,n) = S(n,k)
    }
    const int n = lane & 15;
    const int mbase = ((lane >> 4) & 1) * 8;
#pragma unroll
    for (int j = 0; j < 8; ++j) {
      int m = mbase + j;
      if (m < n) {
        int p = m * (31 - m) / 2 + (n - m - 1);      // np.triu_indices(16, k=1) order
        sX[(size_t)s * XSTR + p] = (__bf16)acc[j];
      }
    }
  }
  __syncthreads();

  // ---- Stage 3: H1 = relu(X @ W1 + b1)  [32x2176 @ 2176x1024] ----
  // 2 Mt x 16 Ngroups(4 tiles) = 32 groups; 4 per wave; A reused across 4 N
  for (int g = wave; g < 32; g += 8) {
    const int mt = g >> 4, ntg = g & 15;
    const __bf16* Abase = sX + (size_t)mt * 16 * XSTR;
    f32x8 acc[4] = {};
    for (int ks = 0; ks < KS1; ++ks) {
      bf16x16 a = load_frag(Abase + ks * 32, XSTR, lane);
#pragma unroll
      for (int j = 0; j < 4; ++j) {
        const int nt = ntg * 4 + j;
        bf16x16 b = load_frag_pk(prm.W1P + (size_t)(nt * KS1 + ks) * 512, lane);
        acc[j] = wmma_bf16(a, b, acc[j]);
      }
    }
    const int mbase = mt * 16 + ((lane >> 4) & 1) * 8;
#pragma unroll
    for (int j = 0; j < 4; ++j) {
      const int n = (ntg * 4 + j) * 16 + (lane & 15);
      const float bias = prm.b1[n];
#pragma unroll
      for (int r = 0; r < 8; ++r)
        sH1[(size_t)(mbase + r) * H1STR + n] = (__bf16)fmaxf(acc[j][r] + bias, 0.0f);
    }
  }
  __syncthreads();

  // ---- Stage 4: H2 = relu(H1 @ W2 + b2)  [32x1024 @ 1024x512] ----
  // 2 Mt x 8 Ngroups = 16 groups; 2 per wave
  for (int g = wave; g < 16; g += 8) {
    const int mt = g >> 3, ntg = g & 7;
    const __bf16* Abase = sH1 + (size_t)mt * 16 * H1STR;
    f32x8 acc[4] = {};
    for (int ks = 0; ks < KS2; ++ks) {
      bf16x16 a = load_frag(Abase + ks * 32, H1STR, lane);
#pragma unroll
      for (int j = 0; j < 4; ++j) {
        const int nt = ntg * 4 + j;
        bf16x16 b = load_frag_pk(prm.W2P + (size_t)(nt * KS2 + ks) * 512, lane);
        acc[j] = wmma_bf16(a, b, acc[j]);
      }
    }
    const int mbase = mt * 16 + ((lane >> 4) & 1) * 8;
#pragma unroll
    for (int j = 0; j < 4; ++j) {
      const int n = (ntg * 4 + j) * 16 + (lane & 15);
      const float bias = prm.b2[n];
#pragma unroll
      for (int r = 0; r < 8; ++r)
        sH2[(size_t)(mbase + r) * H2STR + n] = (__bf16)fmaxf(acc[j][r] + bias, 0.0f);
    }
  }
  __syncthreads();

  // ---- Stage 5: out = relu(H2 @ W3 + b3)  [32x512 @ 512x256] -> global f32 ----
  // 2 Mt x 4 Ngroups = 8 groups; 1 per wave
  {
    const int mt = wave >> 2, ntg = wave & 3;
    const __bf16* Abase = sH2 + (size_t)mt * 16 * H2STR;
    f32x8 acc[4] = {};
    for (int ks = 0; ks < KS3; ++ks) {
      bf16x16 a = load_frag(Abase + ks * 32, H2STR, lane);
#pragma unroll
      for (int j = 0; j < 4; ++j) {
        const int nt = ntg * 4 + j;
        bf16x16 b = load_frag_pk(prm.W3P + (size_t)(nt * KS3 + ks) * 512, lane);
        acc[j] = wmma_bf16(a, b, acc[j]);
      }
    }
    const int mbase = mt * 16 + ((lane >> 4) & 1) * 8;
#pragma unroll
    for (int j = 0; j < 4; ++j) {
      const int n = (ntg * 4 + j) * 16 + (lane & 15);
      const float bias = prm.b3[n];
#pragma unroll
      for (int r = 0; r < 8; ++r)
        prm.out[(size_t)(b0 + mbase + r) * OUTLEN + n] = fmaxf(acc[j][r] + bias, 0.0f);
    }
  }
}

extern "C" void kernel_launch(void* const* d_in, const int* in_sizes, int n_in,
                              void* d_out, int out_size, void* d_ws, size_t ws_size,
                              hipStream_t stream) {
  (void)in_sizes; (void)n_in; (void)out_size; (void)ws_size;
  const float* Wp = (const float*)d_in[16];
  const float* W1 = (const float*)d_in[18];
  const float* W2 = (const float*)d_in[20];
  const float* W3 = (const float*)d_in[22];

  __bf16* ws  = (__bf16*)d_ws;
  __bf16* WpP = ws;                       size_t o = (size_t)8 * KSP * 512;
  __bf16* W1P = ws + o;                   o += (size_t)64 * KS1 * 512;
  __bf16* W2P = ws + o;                   o += (size_t)32 * KS2 * 512;
  __bf16* W3P = ws + o;                   // 16 * KS3 * 512

  // weight prep: bf16 + fragment packing (fully coalesced B streams)
  {
    int total;
    total = 8  * KSP * 512;
    weight_pack_bf16<<<(total + 255) / 256, 256, 0, stream>>>(Wp, WpP, 64,   128,  KSP);
    total = 64 * KS1 * 512;
    weight_pack_bf16<<<(total + 255) / 256, 256, 0, stream>>>(W1, W1P, 2168, 1024, KS1);
    total = 32 * KS2 * 512;
    weight_pack_bf16<<<(total + 255) / 256, 256, 0, stream>>>(W2, W2P, 1024, 512,  KS2);
    total = 16 * KS3 * 512;
    weight_pack_bf16<<<(total + 255) / 256, 256, 0, stream>>>(W3, W3P, 512,  256,  KS3);
  }

  Params prm;
  for (int i = 0; i < 16; ++i) prm.f[i] = (const float*)d_in[i];
  prm.bp = (const float*)d_in[17];
  prm.b1 = (const float*)d_in[19];
  prm.b2 = (const float*)d_in[21];
  prm.b3 = (const float*)d_in[23];
  prm.WpP = WpP; prm.W1P = W1P; prm.W2P = W2P; prm.W3P = W3P;
  prm.out = (float*)d_out;

  const size_t lds_bytes = (size_t)BT * XSTR * 2    // x vectors   (139776 B)
                         + (size_t)BT * H1STR * 2   // hidden 1    ( 66048 B)
                         + (size_t)BT * H2STR * 2;  // hidden 2    ( 33280 B) => ~234 KB
  fused_dlrm<<<32768 / BT, 256, lds_bytes, stream>>>(prm);
}
